// DeepTemplateMatchingModule_81372450390411
// MI455X (gfx1250) — matrix operand order repaired and checked by reading, not compiled, not run on hardware
//
#include <hip/hip_runtime.h>
#include <math.h>

typedef __attribute__((ext_vector_type(16))) _Float16 v16h;
typedef __attribute__((ext_vector_type(8)))  float    v8f;

#define DEV __device__ __forceinline__

// ---------------------------------------------------------------------------
// WMMA helpers (CDNA5 wave32, V_WMMA_F32_16X16X32_F16)
// ---------------------------------------------------------------------------
DEV v8f wmma_f16(v16h a, v16h b, v8f c) {
  return __builtin_amdgcn_wmma_f32_16x16x32_f16(false, a, false, b, (short)0, c,
                                                false, false);
}

// A fragment: 16(M) x 32(K) f16 from row-major A (lda halves).
// lane l: m = mbase + (l&15); half h -> k = kb + 16*(l>>4) + h
DEV v16h frag_a_rm(const _Float16* A, int lda, int mbase, int kb) {
  int lane = threadIdx.x & 31;
  int g = lane >> 4, mm = lane & 15;
  const _Float16* p = A + (size_t)(mbase + mm) * lda + kb + 16 * g;
  v16h r;
#pragma unroll
  for (int i = 0; i < 8; ++i) r[i] = p[i];
#pragma unroll
  for (int i = 0; i < 8; ++i) r[8 + i] = p[8 + i];
  return r;
}

// Same but converting from fp32 source on the fly (GRU hidden state).
DEV v16h frag_a_rm_f32(const float* A, int lda, int mbase, int kb) {
  int lane = threadIdx.x & 31;
  int g = lane >> 4, mm = lane & 15;
  const float* p = A + (size_t)(mbase + mm) * lda + kb + 16 * g;
  v16h r;
#pragma unroll
  for (int i = 0; i < 8; ++i) r[i] = (_Float16)p[i];
#pragma unroll
  for (int i = 0; i < 8; ++i) r[8 + i] = (_Float16)p[8 + i];
  return r;
}

// B fragment: 32(K) x 16(N), logical B[k][n] = W[n*ldk + k]  (W is (N,K) row-major).
// lane l: n = nbase + (l&15); half h -> k = kb + 16*(l>>4) + h   (contiguous!)
DEV v16h frag_b_nk(const _Float16* W, int ldk, int nbase, int kb) {
  int lane = threadIdx.x & 31;
  int g = lane >> 4, nn = lane & 15;
  const _Float16* p = W + (size_t)(nbase + nn) * ldk + kb + 16 * g;
  v16h r;
#pragma unroll
  for (int i = 0; i < 16; ++i) r[i] = p[i];
  return r;
}

// B fragment from (K,N) row-major matrix (strided gather).
DEV v16h frag_b_kn(const _Float16* B, int ldn, int nbase, int kb) {
  int lane = threadIdx.x & 31;
  int g = lane >> 4, nn = lane & 15;
  const _Float16* p = B + (size_t)(kb + 16 * g) * ldn + nbase + nn;
  v16h r;
#pragma unroll
  for (int i = 0; i < 16; ++i) r[i] = p[(size_t)i * ldn];
  return r;
}

DEV float sigmoidf_(float x) { return 1.f / (1.f + expf(-x)); }

// ---------------------------------------------------------------------------
// fp32 -> f16 convert with K->KP zero padding per row (KP==K => plain convert)
// ---------------------------------------------------------------------------
__global__ void cvt_pad_f16_kernel(const float* __restrict__ s,
                                   _Float16* __restrict__ d, int K, int KP,
                                   int total) {
  int i = blockIdx.x * blockDim.x + threadIdx.x;
  if (i >= total) return;
  int row = i / KP, k = i - row * KP;
  d[i] = (k < K) ? (_Float16)s[(size_t)row * K + k] : (_Float16)0.f;
}

// ---------------------------------------------------------------------------
// conv1: 1->16 channels, 5x5 VALID, direct fp32, f16 output. x:(4,128,W0)
// ---------------------------------------------------------------------------
__global__ void conv1_kernel(const float* __restrict__ x,
                             const float* __restrict__ w,
                             const float* __restrict__ bias,
                             _Float16* __restrict__ out, int W0) {
  int Wo = W0 - 4;
  size_t idx = (size_t)blockIdx.x * blockDim.x + threadIdx.x;
  size_t total = (size_t)4 * 16 * 124 * Wo;
  if (idx >= total) return;
  int xo = (int)(idx % Wo);
  size_t t = idx / Wo;
  int yo = (int)(t % 124); t /= 124;
  int co = (int)(t % 16);
  int b = (int)(t / 16);
  float acc = bias[co];
  const float* xp = x + ((size_t)b * 128 + yo) * W0 + xo;
  const float* wp = w + co * 25;
#pragma unroll
  for (int dy = 0; dy < 5; ++dy)
#pragma unroll
    for (int dx = 0; dx < 5; ++dx)
      acc += xp[(size_t)dy * W0 + dx] * wp[dy * 5 + dx];
  out[idx] = (_Float16)acc;
}

// ---------------------------------------------------------------------------
// Implicit-GEMM conv via WMMA with LDS-staged activations.
// in:(4,CIN,Hi,Wi) f16, wh:(COUT,KP) f16 zero-padded, out:(4,COUT,Ho,Wo) f16.
// Block = 256 thr = 8 waves = (COUT/16) co-tiles x NXT x-tiles.
// Per block: stage CIN x 5 rows x (NXT*16+4) input strip into LDS once; every
// wave builds B fragments from LDS (shared across co-tiles), A from global
// (weights, L0/L2 resident). M=16 out-channels, N=16 out-x, K=CIN*25 (padded).
// ---------------------------------------------------------------------------
template <int CIN, int COUT, int NXT>
__global__ void __launch_bounds__(256) conv_wmma_kernel(
    const _Float16* __restrict__ in, const _Float16* __restrict__ wh,
    const float* __restrict__ bias, _Float16* __restrict__ out, int Hi, int Wi) {
  const int K = CIN * 25;
  const int KP = (K + 31) & ~31;
  const int NCOT = COUT / 16;
  const int XW = NXT * 16 + 4;           // staged strip width (incl. halo)
  __shared__ _Float16 sact[CIN * 5 * XW];
  int Ho = Hi - 4, Wo = Wi - 4;
  int y = blockIdx.y;
  int b = blockIdx.z;
  int xb0 = blockIdx.x * (NXT * 16);
  const _Float16* inb = in + (size_t)b * CIN * Hi * Wi;
  // cooperative stage (all waves participate before any early exit)
  for (int i = threadIdx.x; i < CIN * 5 * XW; i += 256) {
    int x = i % XW;
    int t = i / XW;
    int dy = t % 5;
    int ci = t / 5;
    int sx = xb0 + x;
    if (sx > Wi - 1) sx = Wi - 1;  // clamp: edge blocks stay in-bounds
    sact[i] = inb[((size_t)ci * Hi + (y + dy)) * Wi + sx];
  }
  __syncthreads();
  int wave = threadIdx.x >> 5;
  int cot = wave % NCOT;
  int xt = wave / NCOT;
  int x0 = xb0 + xt * 16;
  if (x0 >= Wo) return;  // wave-uniform exit (EXEC all-ones for WMMA)
  int lane = threadIdx.x & 31;
  int g = lane >> 4, nn = lane & 15;
  int xl = xt * 16 + nn;  // local x inside staged strip (xl+dx <= XW-1 always)
  v8f c = {};
  for (int kb = 0; kb < KP; kb += 32) {
    v16h a = frag_a_rm(wh, KP, cot * 16, kb);
    v16h bf;
#pragma unroll
    for (int h = 0; h < 16; ++h) {
      int k = kb + (g << 4) + h;
      _Float16 v = (_Float16)0.f;
      if (k < K) {
        int ci = k / 25, rr = k - ci * 25;
        int dy = rr / 5, dx = rr - dy * 5;
        v = sact[(ci * 5 + dy) * XW + xl + dx];
      }
      bf[h] = v;
    }
    c = wmma_f16(a, bf, c);
  }
  int xo = x0 + nn;
  if (xo < Wo) {
    size_t ob = (size_t)b * COUT * Ho * Wo;
#pragma unroll
    for (int r = 0; r < 8; ++r) {
      int co = cot * 16 + r + 8 * g;
      out[ob + ((size_t)co * Ho + y) * Wo + xo] = (_Float16)(c[r] + bias[co]);
    }
  }
}

// ---------------------------------------------------------------------------
// maxpool (2,5)/(2,1): (4,64,116,Wc) f16 -> (4,64,58,Wc-4) f16 flat
// ---------------------------------------------------------------------------
__global__ void maxpool_kernel(const _Float16* __restrict__ in,
                               _Float16* __restrict__ out, int Wc) {
  int Wp = Wc - 4;
  size_t idx = (size_t)blockIdx.x * blockDim.x + threadIdx.x;
  size_t total = (size_t)4 * 64 * 58 * Wp;
  if (idx >= total) return;
  int w = (int)(idx % Wp);
  size_t t = idx / Wp;
  int h2 = (int)(t % 58); t /= 58;
  int c = (int)(t % 64);
  int b = (int)(t / 64);
  const _Float16* p = in + (((size_t)b * 64 + c) * 116 + 2 * h2) * Wc + w;
  float m = -3.4e38f;
#pragma unroll
  for (int dy = 0; dy < 2; ++dy)
#pragma unroll
    for (int dx = 0; dx < 5; ++dx)
      m = fmaxf(m, (float)p[(size_t)dy * Wc + dx]);
  out[idx] = (_Float16)m;
}

// ---------------------------------------------------------------------------
// Generic WMMA GEMM: C[m][n] = sum_k A[m][k]*W[n][k] (+bias[n]).
// A f16 (M,K) lda=K; W f16 (N,K); optional batching via grid.y + strides.
// 256 threads = 8 waves, one 16x16 tile each. M%16==0, N%16==0, K%32==0.
// ---------------------------------------------------------------------------
template <bool HALF_OUT, bool HAS_BIAS>
__global__ void __launch_bounds__(256) gemm_kernel(
    const _Float16* __restrict__ A, const _Float16* __restrict__ W,
    const float* __restrict__ bias, void* __restrict__ C, int M, int N, int K,
    size_t sA, size_t sW, size_t sC) {
  int bidx = blockIdx.y;
  const _Float16* Ab = A + (size_t)bidx * sA;
  const _Float16* Wb = W + (size_t)bidx * sW;
  int wave = threadIdx.x >> 5;
  int tile = blockIdx.x * 8 + wave;
  int nt = N >> 4;
  int total = (M >> 4) * nt;
  if (tile >= total) return;
  int mbase = (tile / nt) * 16;
  int nbase = (tile % nt) * 16;
  v8f c = {};
  for (int kb = 0; kb < K; kb += 32) {
    v16h a = frag_a_rm(Ab, K, mbase, kb);
    v16h bb = frag_b_nk(Wb, K, nbase, kb);
    c = wmma_f16(a, bb, c);
  }
  int lane = threadIdx.x & 31;
  int g = lane >> 4, nn = lane & 15;
  int n = nbase + nn;
  float bv = 0.f;
  if (HAS_BIAS) bv = bias[n];
#pragma unroll
  for (int r = 0; r < 8; ++r) {
    int m = mbase + r + 8 * g;
    float v = c[r] + bv;
    if (HALF_OUT)
      ((_Float16*)C)[(size_t)bidx * sC + (size_t)m * N + n] = (_Float16)v;
    else
      ((float*)C)[(size_t)bidx * sC + (size_t)m * N + n] = v;
  }
}

// ---------------------------------------------------------------------------
// One GRU step: gh = hprev @ whh^T + bhh (WMMA -> LDS), then fused gates.
// 384 threads = 12 waves; block handles 16 rows x all 192 gate columns.
// hprev == nullptr for t==0 (h0 = 0 -> gh = bhh).
// ---------------------------------------------------------------------------
__global__ void __launch_bounds__(384) gru_step_kernel(
    const float* __restrict__ GIt, const float* __restrict__ hprev,
    const _Float16* __restrict__ whh, const float* __restrict__ bhh,
    float* __restrict__ hnew) {
  __shared__ float gh[16][200];  // 192 + pad
  int wave = threadIdx.x >> 5;   // 0..11
  int mbase = blockIdx.x * 16;
  int nbase = wave * 16;
  int lane = threadIdx.x & 31;
  int g = lane >> 4, nn = lane & 15;
  v8f c = {};
  if (hprev) {
    for (int kb = 0; kb < 64; kb += 32) {
      v16h a = frag_a_rm_f32(hprev, 64, mbase, kb);
      v16h bb = frag_b_nk(whh, 64, nbase, kb);
      c = wmma_f16(a, bb, c);
    }
  }
  float bv = bhh[nbase + nn];
#pragma unroll
  for (int r = 0; r < 8; ++r) gh[r + 8 * g][nbase + nn] = c[r] + bv;
  __syncthreads();
  for (int idx = threadIdx.x; idx < 16 * 64; idx += 384) {
    int mm = idx >> 6, j = idx & 63;
    size_t row = (size_t)mbase + mm;
    const float* gi = GIt + row * 192;
    float rr = sigmoidf_(gi[j] + gh[mm][j]);
    float zz = sigmoidf_(gi[64 + j] + gh[mm][64 + j]);
    float nl = tanhf(gi[128 + j] + rr * gh[mm][128 + j]);
    float hp = hprev ? hprev[row * 64 + j] : 0.f;
    hnew[row * 64 + j] = (1.f - zz) * nl + zz * hp;
  }
}

// ---------------------------------------------------------------------------
// Row softmax of S (4,T,Ne) fp32, written transposed as f16: PT[b][e][t].
// ---------------------------------------------------------------------------
__global__ void __launch_bounds__(256) softmax_transpose_kernel(
    const float* __restrict__ S, _Float16* __restrict__ PT, int T, int Ne) {
  int row = blockIdx.x;
  int b = row / T, t = row - b * T;
  const float* s = S + ((size_t)b * T + t) * Ne;
  __shared__ float scr[256];
  int tid = threadIdx.x;
  float mx = -3.4e38f;
  for (int e = tid; e < Ne; e += 256) mx = fmaxf(mx, s[e]);
  scr[tid] = mx;
  __syncthreads();
  for (int o = 128; o > 0; o >>= 1) {
    if (tid < o) scr[tid] = fmaxf(scr[tid], scr[tid + o]);
    __syncthreads();
  }
  mx = scr[0];
  __syncthreads();
  float sum = 0.f;
  for (int e = tid; e < Ne; e += 256) sum += expf(s[e] - mx);
  scr[tid] = sum;
  __syncthreads();
  for (int o = 128; o > 0; o >>= 1) {
    if (tid < o) scr[tid] += scr[tid + o];
    __syncthreads();
  }
  float inv = 1.f / scr[0];
  for (int e = tid; e < Ne; e += 256)
    PT[((size_t)b * Ne + e) * T + t] = (_Float16)(expf(s[e] - mx) * inv);
}

// ---------------------------------------------------------------------------
// context: TT[b][e][d] = | sum_t PT[b][e][t]*TH[b][t][d]  -  E[b][e][d] |
// ---------------------------------------------------------------------------
__global__ void __launch_bounds__(256) context_absdiff_kernel(
    const _Float16* __restrict__ PT, const _Float16* __restrict__ TH,
    const float* __restrict__ E, float* __restrict__ TT, int Ne, int Kt) {
  int b = blockIdx.y;
  const _Float16* Ab = PT + (size_t)b * Ne * Kt;
  const _Float16* Bb = TH + (size_t)b * Kt * 64;
  int wave = threadIdx.x >> 5;
  int tile = blockIdx.x * 8 + wave;
  int total = (Ne >> 4) * 4;
  if (tile >= total) return;
  int mbase = (tile >> 2) * 16;
  int nbase = (tile & 3) * 16;
  v8f c = {};
  for (int kb = 0; kb < Kt; kb += 32) {
    v16h a = frag_a_rm(Ab, Kt, mbase, kb);
    v16h bb = frag_b_kn(Bb, 64, nbase, kb);
    c = wmma_f16(a, bb, c);
  }
  int lane = threadIdx.x & 31;
  int g = lane >> 4, nn = lane & 15;
  int n = nbase + nn;
#pragma unroll
  for (int r = 0; r < 8; ++r) {
    int m = mbase + r + 8 * g;
    size_t o = ((size_t)b * Ne + m) * 64 + n;
    TT[o] = fabsf(c[r] - E[o]);
  }
}

// ---------------------------------------------------------------------------
// att = softmax_e(E@a_w^T + a_b); red[b][d] = sum_e TT[b][e][d]*att[e]
// ---------------------------------------------------------------------------
__global__ void __launch_bounds__(256) attn_reduce_kernel(
    const float* __restrict__ E, const float* __restrict__ aw,
    const float* __restrict__ ab, const float* __restrict__ TT,
    float* __restrict__ red, int Ne) {
  int b = blockIdx.x;
  int tid = threadIdx.x;
  __shared__ float att[2048];
  __shared__ float scr[256];
  const float* Eb = E + (size_t)b * Ne * 64;
  float mx = -3.4e38f;
  for (int e = tid; e < Ne; e += 256) {
    const float* ev = Eb + (size_t)e * 64;
    float l = ab[0];
#pragma unroll
    for (int d = 0; d < 64; ++d) l += ev[d] * aw[d];
    att[e] = l;
    mx = fmaxf(mx, l);
  }
  scr[tid] = mx;
  __syncthreads();
  for (int o = 128; o > 0; o >>= 1) {
    if (tid < o) scr[tid] = fmaxf(scr[tid], scr[tid + o]);
    __syncthreads();
  }
  mx = scr[0];
  __syncthreads();
  float sum = 0.f;
  for (int e = tid; e < Ne; e += 256) {
    float v = expf(att[e] - mx);
    att[e] = v;
    sum += v;
  }
  scr[tid] = sum;
  __syncthreads();
  for (int o = 128; o > 0; o >>= 1) {
    if (tid < o) scr[tid] += scr[tid + o];
    __syncthreads();
  }
  float inv = 1.f / scr[0];
  __syncthreads();
  for (int e = tid; e < Ne; e += 256) att[e] *= inv;
  __syncthreads();
  int d = tid & 63, st = tid >> 6;
  float acc = 0.f;
  for (int e = st; e < Ne; e += 4) acc += TT[((size_t)b * Ne + e) * 64 + d] * att[e];
  scr[tid] = acc;
  __syncthreads();
  if (tid < 64)
    red[b * 64 + tid] = scr[tid] + scr[64 + tid] + scr[128 + tid] + scr[192 + tid];
}

// ---------------------------------------------------------------------------
// h = relu(red@h_w^T + h_b); out = softmax(h@c_w^T + c_b)   (tiny)
// ---------------------------------------------------------------------------
__global__ void __launch_bounds__(128) head_kernel(
    const float* __restrict__ red, const float* __restrict__ hw,
    const float* __restrict__ hb, const float* __restrict__ cw,
    const float* __restrict__ cb, float* __restrict__ out) {
  __shared__ float h[128];
  int tid = threadIdx.x;
  for (int b = 0; b < 4; ++b) {
    const float* rb = red + b * 64;
    float v = hb[tid];
#pragma unroll
    for (int d = 0; d < 64; ++d) v += rb[d] * hw[tid * 64 + d];
    h[tid] = fmaxf(v, 0.f);
    __syncthreads();
    if (tid == 0) {
      float l0 = cb[0], l1 = cb[1];
      for (int j = 0; j < 128; ++j) {
        l0 += h[j] * cw[j];
        l1 += h[j] * cw[128 + j];
      }
      float m = fmaxf(l0, l1);
      float e0 = expf(l0 - m), e1 = expf(l1 - m);
      out[b * 2 + 0] = e0 / (e0 + e1);
      out[b * 2 + 1] = e1 / (e0 + e1);
    }
    __syncthreads();
  }
}

// ---------------------------------------------------------------------------
// host orchestration
// ---------------------------------------------------------------------------
extern "C" void kernel_launch(void* const* d_in, const int* in_sizes, int n_in,
                              void* d_out, int out_size, void* d_ws,
                              size_t ws_size, hipStream_t stream) {
  (void)in_sizes; (void)n_in; (void)out_size;
  auto in = [&](int i) { return (const float*)d_in[i]; };

  // ---- workspace carve (two big reused arenas + small persistent buffers)
  char* p = (char*)d_ws;
  auto take = [&](size_t n) -> char* {
    char* r = p;
    p += (n + 255) & ~(size_t)255;
    return r;
  };
  const size_t SZ_BUFA = (size_t)4 * 64 * 116 * 2036 * 2;  // >= c1_e, c3_*, S
  const size_t SZ_BUFB = (size_t)4 * 32 * 120 * 2040 * 2;  // >= c2_*, pooled_*, PT
  _Float16* bufA = (_Float16*)take(SZ_BUFA);
  _Float16* bufB = (_Float16*)take(SZ_BUFB);
  _Float16* XF = (_Float16*)take((size_t)4 * 2032 * 64 * 2);
  float* GI = (float*)take((size_t)4 * 2032 * 192 * 4);
  float* ysE = (float*)take((size_t)4 * 2032 * 64 * 4);
  float* ysT = (float*)take((size_t)4 * 1024 * 64 * 4);
  _Float16* EH = (_Float16*)take((size_t)4 * 2032 * 64 * 2);
  _Float16* TH = (_Float16*)take((size_t)4 * 1024 * 64 * 2);
  float* TT = (float*)take((size_t)4 * 2032 * 64 * 4);
  float* red = (float*)take(4 * 64 * 4);
  _Float16* w2h = (_Float16*)take((size_t)32 * 416 * 2);
  _Float16* w3h = (_Float16*)take((size_t)64 * 800 * 2);
  _Float16* wlh = (_Float16*)take((size_t)64 * 3712 * 2);
  _Float16* wihh = (_Float16*)take((size_t)192 * 64 * 2);
  _Float16* whhh = (_Float16*)take((size_t)192 * 64 * 2);
  if ((size_t)(p - (char*)d_ws) > ws_size) return;  // insufficient scratch

  auto cvt = [&](const float* s, _Float16* d, int K, int KP, int total) {
    cvt_pad_f16_kernel<<<dim3((total + 255) / 256), 256, 0, stream>>>(s, d, K, KP, total);
  };
  auto gemm = [&](const _Float16* A, const _Float16* W, const float* bias, void* C,
                  int M, int N, int K, int batch, size_t sA, size_t sW, size_t sC,
                  bool halfOut) {
    int tiles = (M / 16) * (N / 16);
    dim3 grid((tiles + 7) / 8, batch);
    if (halfOut) {
      if (bias) gemm_kernel<true, true><<<grid, 256, 0, stream>>>(A, W, bias, C, M, N, K, sA, sW, sC);
      else      gemm_kernel<true, false><<<grid, 256, 0, stream>>>(A, W, bias, C, M, N, K, sA, sW, sC);
    } else {
      if (bias) gemm_kernel<false, true><<<grid, 256, 0, stream>>>(A, W, bias, C, M, N, K, sA, sW, sC);
      else      gemm_kernel<false, false><<<grid, 256, 0, stream>>>(A, W, bias, C, M, N, K, sA, sW, sC);
    }
  };

  auto run_branch = [&](const float* x, int W0, int Wp, const float* w1,
                        const float* b1, const float* w2, const float* b2,
                        const float* w3, const float* b3, const float* lw,
                        const float* lb, const float* wih, const float* whh,
                        const float* bih, const float* bhh, float* ys) {
    int W1 = W0 - 4, W2 = W0 - 8, W3 = W0 - 12;
    // conv1 (direct fp32 -> f16)
    size_t n1 = (size_t)4 * 16 * 124 * W1;
    conv1_kernel<<<dim3((unsigned)((n1 + 255) / 256)), 256, 0, stream>>>(x, w1, b1, bufA, W0);
    // conv2: implicit GEMM (K=400 padded to 416), LDS-staged activations
    cvt(w2, w2h, 400, 416, 32 * 416);
    conv_wmma_kernel<16, 32, 4><<<dim3((W2 + 63) / 64, 120, 4), 256, 0, stream>>>(
        bufA, w2h, b2, bufB, 124, W1);
    // conv3: implicit GEMM (K=800), LDS-staged activations
    cvt(w3, w3h, 800, 800, 64 * 800);
    conv_wmma_kernel<32, 64, 2><<<dim3((W3 + 31) / 32, 116, 4), 256, 0, stream>>>(
        bufB, w3h, b3, bufA, 120, W2);
    // maxpool -> pooled flat f16 (= reshape rows of length 3712)
    size_t np = (size_t)4 * 64 * 58 * Wp;
    maxpool_kernel<<<dim3((unsigned)((np + 255) / 256)), 256, 0, stream>>>(bufA, bufB, W3);
    // linear 3712->64 (M = 4*Wp rows)
    cvt(lw, wlh, 3712, 3712, 64 * 3712);
    gemm(bufB, wlh, lb, XF, 4 * Wp, 64, 3712, 1, 0, 0, 0, true);
    // GRU input gates for all 4 steps at once: GI = X @ w_ih^T + b_ih
    cvt(wih, wihh, 64, 64, 192 * 64);
    gemm(XF, wihh, bih, GI, 4 * Wp, 192, 64, 1, 0, 0, 0, false);
    cvt(whh, whhh, 64, 64, 192 * 64);
    // sequential recurrence, 4 steps, h stored in-place in ys
    for (int t = 0; t < 4; ++t)
      gru_step_kernel<<<dim3(Wp / 16), 384, 0, stream>>>(
          GI + (size_t)t * Wp * 192,
          t ? ys + (size_t)(t - 1) * Wp * 64 : nullptr, whhh, bhh,
          ys + (size_t)t * Wp * 64);
  };

  // ---- evaluation branch (W0=2048, Wp=2032), template branch (W0=1040, Wp=1024)
  run_branch(in(0), 2048, 2032, in(2), in(3), in(4), in(5), in(6), in(7), in(8),
             in(9), in(10), in(11), in(12), in(13), ysE);
  run_branch(in(1), 1040, 1024, in(14), in(15), in(16), in(17), in(18), in(19),
             in(20), in(21), in(22), in(23), in(24), in(25), ysT);

  // ---- head
  cvt(ysE, EH, 4 * 2032 * 64, 4 * 2032 * 64, 4 * 2032 * 64);
  cvt(ysT, TH, 4 * 1024 * 64, 4 * 1024 * 64, 4 * 1024 * 64);
  // scores S[b][t][e] = T . E   (batched, fp32 out into bufA)
  gemm(TH, EH, nullptr, (float*)bufA, 1024, 2032, 64, 4, (size_t)1024 * 64,
       (size_t)2032 * 64, (size_t)1024 * 2032, false);
  // softmax over e, stored transposed (e-major) as f16 Pᵀ into bufB
  softmax_transpose_kernel<<<dim3(4 * 1024), 256, 0, stream>>>((float*)bufA, bufB, 1024, 2032);
  // tt = |Pᵀ @ T - E|
  context_absdiff_kernel<<<dim3(64, 4), 256, 0, stream>>>(bufB, TH, ysE, TT, 2032, 1024);
  // attention softmax + weighted reduction
  attn_reduce_kernel<<<dim3(4), 256, 0, stream>>>(ysE, in(26), in(27), TT, red, 2032);
  // dense head + final softmax
  head_kernel<<<dim3(1), 128, 0, stream>>>(red, in(28), in(29), in(30), in(31),
                                           (float*)d_out);
}
